// AttrFusion_69054484185646
// MI455X (gfx1250) — matrix operand order repaired and checked by reading, not compiled, not run on hardware
//
#include <hip/hip_runtime.h>
#include <hip/hip_bf16.h>

// ---------------------------------------------------------------------------
// AttrFusion on MI455X (gfx1250):
//  - bilinear resize (align_corners) of attr_feat precomputed per level into
//    bf16 [b][pix][ch2048] workspace tensors (bandwidth pass, ~145 MB total)
//  - grouped 3x3 conv as bf16 implicit-GEMM on v_wmma_f32_16x16x32_bf16;
//    im2col A-tiles double-buffered in LDS via GLOBAL_LOAD_ASYNC_TO_LDS_B128
//    (ASYNCcnt in-order completion => s_wait_asynccnt 5 overlaps DMA of the
//    next tile with WMMA on the current one)
//  - BN + ReLU + softmax-attention group reduction + residual fused in the
//    epilogue. f32 accumulation throughout.
// ---------------------------------------------------------------------------

typedef __attribute__((ext_vector_type(16))) __bf16 v16bf;
typedef __attribute__((ext_vector_type(8)))  float  v8f;
typedef __attribute__((ext_vector_type(8)))  unsigned short u16x8;
typedef __attribute__((ext_vector_type(16))) unsigned short u16x16;

#define K_CLS 8
#define C_INC 256
#define ACH   2048          // K_CLS * C_INC
#define AH    64
#define AWD   64
#define APIX  4096          // AH*AWD
#define EPSV  1e-5f

// ---- workspace layout (bytes). Requires ws_size >= ~216 MiB. ----
#define WS_ATTRT   ((size_t)0)            // bf16 attr [b][pix][ch] (= resized lvl1)  33,554,432
#define WS_RES0    ((size_t)33554432)     // bf16 resized lvl0 [b][16384][2048]      134,217,728
#define WS_RES2    ((size_t)167772160)    // bf16 resized lvl2                         8,388,608
#define WS_RES3    ((size_t)176160768)    // bf16 resized lvl3                         2,097,152
#define WS_RES4    ((size_t)178257920)    // bf16 resized lvl4                           524,288
#define WS_WFRAG   ((size_t)178782208)    // bf16 weight fragments                    47,185,920
#define WS_AVGATTR ((size_t)225968128)    // f32 [b][2048]
#define WS_AM      ((size_t)225984512)    // f32 [b*8+k][256]
#define WS_AVGX    ((size_t)226000896)    // f32 [lvl*2+b][256]
#define WS_FX      ((size_t)226011136)    // f32 [lvl*2+b][256]
#define WS_ATTN    ((size_t)226021376)    // f32 [lvl*2+b][8]
#define WS_BNS     ((size_t)226021888)    // f32 [lvl][2048]
#define WS_BNB     ((size_t)226062848)    // f32 [lvl][2048]
#define WS_ZERO    ((size_t)226103808)    // 256 B zero page (padding source)

static __device__ __forceinline__ float bfbits2f(unsigned short h) {
    unsigned u = ((unsigned)h) << 16;
    return __builtin_bit_cast(float, u);
}
static __device__ __forceinline__ unsigned short f2bfbits(float f) {
    unsigned u = __builtin_bit_cast(unsigned, f);
    unsigned r = u + 0x7FFFu + ((u >> 16) & 1u);   // round-to-nearest-even
    return (unsigned short)(r >> 16);
}

// ---------------------------------------------------------------------------
// K0: attr_feat f32 [b][ch][pix] -> bf16 transposed [b][pix][ch]
// ---------------------------------------------------------------------------
__global__ void k_attr_to_bf16_t(const float* __restrict__ attr,
                                 unsigned short* __restrict__ attrT) {
    size_t i = (size_t)blockIdx.x * 256 + threadIdx.x;    // 2*2048*4096 total
    int pix = (int)(i & 4095);
    size_t t = i >> 12;
    int ch = (int)(t & 2047);
    int b  = (int)(t >> 11);
    attrT[(((size_t)b * APIX + pix) * ACH) + ch] = f2bfbits(attr[i]);
}

// ---------------------------------------------------------------------------
// K0b: bilinear resize (align_corners=True) attrT -> bf16 [b][HW][2048]
// ---------------------------------------------------------------------------
__global__ void k_resize(const unsigned short* __restrict__ src,  // [b][4096][2048]
                         unsigned short* __restrict__ dst,        // [b][HW][2048]
                         int H, int Wd, int shiftW, int HW) {
    int i = blockIdx.x * 256 + threadIdx.x;   // total = 2*HW*256
    int c8  = i & 255;
    int rest = i >> 8;
    int pix = rest % HW;
    int b   = rest / HW;
    int y = pix >> shiftW, x = pix & (Wd - 1);
    float sy = (float)(AH - 1) / (float)(H - 1);
    float sx = (float)(AWD - 1) / (float)(Wd - 1);
    float fy = y * sy, fx = x * sx;
    int y0 = (int)fy, x0 = (int)fx;
    int y1 = min(y0 + 1, AH - 1), x1 = min(x0 + 1, AWD - 1);
    float wy = fy - (float)y0, wx = fx - (float)x0;
    float w00 = (1.f - wx) * (1.f - wy), w01 = wx * (1.f - wy);
    float w10 = (1.f - wx) * wy,         w11 = wx * wy;
    const unsigned short* base = src + (size_t)b * APIX * ACH + c8 * 8;
    u16x8 a00 = *(const u16x8*)(base + (size_t)(y0 * AWD + x0) * ACH);
    u16x8 a01 = *(const u16x8*)(base + (size_t)(y0 * AWD + x1) * ACH);
    u16x8 a10 = *(const u16x8*)(base + (size_t)(y1 * AWD + x0) * ACH);
    u16x8 a11 = *(const u16x8*)(base + (size_t)(y1 * AWD + x1) * ACH);
    u16x8 r;
#pragma unroll
    for (int e = 0; e < 8; ++e) {
        float v = w00 * bfbits2f(a00[e]) + w01 * bfbits2f(a01[e]) +
                  w10 * bfbits2f(a10[e]) + w11 * bfbits2f(a11[e]);
        r[e] = f2bfbits(v);
    }
    *(u16x8*)(dst + ((size_t)b * HW + pix) * ACH + c8 * 8) = r;
}

// ---------------------------------------------------------------------------
// K1: conv_w f32 [lvl][o=g*256+n][i][ky][kx] -> bf16 per-lane WMMA B-fragments
//  layout: [lvl][g][tap][cchunk(8)][nchunk(16)][lane(32)][elem(16)]
//  lane l: column n=l&15, K rows 16*(e>>3)+8*(l>>4)+(e&7)   (B 32x16 fragment, K = in-ch)
// ---------------------------------------------------------------------------
__global__ void k_weights_frag(const float* __restrict__ w,
                               unsigned short* __restrict__ wf) {
    size_t i = (size_t)blockIdx.x * 256 + threadIdx.x;    // 23,592,960 total
    int e  = (int)(i & 15);
    int l  = (int)((i >> 4) & 31);
    int nc = (int)((i >> 9) & 15);
    int cc = (int)((i >> 13) & 7);
    size_t r = i >> 16;
    int t = (int)(r % 9); r /= 9;
    int g = (int)(r & 7);
    int lvl = (int)(r >> 3);
    int n = l & 15;
    int k = ((e >> 3) << 4) + ((l >> 4) << 3) + (e & 7);
    int o  = g * 256 + nc * 16 + n;
    int ci = cc * 32 + k;
    int ky = t / 3, kx = t - ky * 3;
    size_t src = ((((size_t)lvl * ACH + o) * C_INC + ci) * 3 + ky) * 3 + kx;
    wf[i] = f2bfbits(w[src]);
}

// ---------------------------------------------------------------------------
// K2: global average pool of attr_feat over 4096 pixels per (b, ch)
// ---------------------------------------------------------------------------
__global__ void k_avg_attr(const float* __restrict__ attr, float* __restrict__ avg) {
    __shared__ float red[256];
    const float* p = attr + (size_t)blockIdx.x * APIX;
    float s = 0.f;
    for (int i = threadIdx.x; i < APIX; i += 256) s += p[i];
    red[threadIdx.x] = s; __syncthreads();
    for (int st = 128; st > 0; st >>= 1) {
        if (threadIdx.x < st) red[threadIdx.x] += red[threadIdx.x + st];
        __syncthreads();
    }
    if (threadIdx.x == 0) avg[blockIdx.x] = red[0] * (1.0f / APIX);
}

// ---------------------------------------------------------------------------
// K3: tiny MLP  out[row][co] = dot(in[row], W[co]) + bias[co]
// ---------------------------------------------------------------------------
__global__ void k_mlp(const float* __restrict__ in, const float* __restrict__ W,
                      const float* __restrict__ bias, float* __restrict__ out) {
    __shared__ float red[256];
    int co  = blockIdx.x & 255;
    int row = blockIdx.x >> 8;
    float v = in[row * 256 + threadIdx.x] * W[co * 256 + threadIdx.x];
    red[threadIdx.x] = v; __syncthreads();
    for (int st = 128; st > 0; st >>= 1) {
        if (threadIdx.x < st) red[threadIdx.x] += red[threadIdx.x + st];
        __syncthreads();
    }
    if (threadIdx.x == 0) out[row * 256 + co] = red[0] + bias[co];
}

// ---------------------------------------------------------------------------
// K4: per-level per-(b,c) average pool of a feature map
// ---------------------------------------------------------------------------
__global__ void k_avg_feat(const float* __restrict__ x, float* __restrict__ avg, int HW) {
    __shared__ float red[256];
    const float* p = x + (size_t)blockIdx.x * HW;
    float s = 0.f;
    for (int i = threadIdx.x; i < HW; i += 256) s += p[i];
    red[threadIdx.x] = s; __syncthreads();
    for (int st = 128; st > 0; st >>= 1) {
        if (threadIdx.x < st) red[threadIdx.x] += red[threadIdx.x + st];
        __syncthreads();
    }
    if (threadIdx.x == 0) avg[blockIdx.x] = red[0] / (float)HW;
}

// ---------------------------------------------------------------------------
// K5: logits[b,k] = dot(am[b,k,:], fx[lvl,b,:]); softmax over k -> attn
// ---------------------------------------------------------------------------
__global__ void k_attn(const float* __restrict__ am, const float* __restrict__ fx,
                       float* __restrict__ attn) {
    __shared__ float red[256];
    __shared__ float lg[K_CLS];
    int lb = blockIdx.x;
    int b  = lb & 1;
    const float* f = fx + (size_t)lb * 256;
    for (int k = 0; k < K_CLS; ++k) {
        float v = am[((size_t)(b * K_CLS + k)) * 256 + threadIdx.x] * f[threadIdx.x];
        red[threadIdx.x] = v; __syncthreads();
        for (int st = 128; st > 0; st >>= 1) {
            if (threadIdx.x < st) red[threadIdx.x] += red[threadIdx.x + st];
            __syncthreads();
        }
        if (threadIdx.x == 0) lg[k] = red[0];
        __syncthreads();
    }
    if (threadIdx.x == 0) {
        float m = lg[0];
        for (int k = 1; k < K_CLS; ++k) m = fmaxf(m, lg[k]);
        float e[K_CLS], s = 0.f;
        for (int k = 0; k < K_CLS; ++k) { e[k] = __expf(lg[k] - m); s += e[k]; }
        float inv = 1.0f / s;
        for (int k = 0; k < K_CLS; ++k) attn[lb * K_CLS + k] = e[k] * inv;
    }
}

// ---------------------------------------------------------------------------
// K6: fold BN into scale/shift per (lvl, out-channel); also clear zero page
// ---------------------------------------------------------------------------
__global__ void k_bnfold(const float* __restrict__ gamma, const float* __restrict__ beta,
                         const float* __restrict__ mean, const float* __restrict__ var,
                         float* __restrict__ scale, float* __restrict__ shift,
                         float* __restrict__ zpage) {
    if (blockIdx.x == 0 && threadIdx.x < 64) zpage[threadIdx.x] = 0.f;
    int i = blockIdx.x * 256 + threadIdx.x;
    if (i >= 5 * ACH) return;
    float sc = gamma[i] * rsqrtf(var[i] + EPSV);
    scale[i] = sc;
    shift[i] = beta[i] - mean[i] * sc;
}

// ---------------------------------------------------------------------------
// K7: grouped-conv (bf16 WMMA) + BN + ReLU + attn-weighted group sum +
//     residual + ReLU, reading the pre-resized bf16 tensor.
//  grid = (HW/32, B); block = 256 (8 wave32, 2Mx4N); wave tile = 16M x 64N
//  (4 accumulators -> one A fragment feeds 4 WMMAs). A tiles double-buffered
//  in LDS via GLOBAL_LOAD_ASYNC_TO_LDS_B128; exactly 5 async issues per wave
//  per tile (padded to 1280 xfers) so s_wait_asynccnt 5 == "previous tile
//  landed" while the next tile's DMA is still in flight.
// ---------------------------------------------------------------------------
__global__ __launch_bounds__(256) void k_fused_conv(
    const unsigned short* __restrict__ res,     // bf16 [b][HW][2048] resized attr
    const unsigned short* __restrict__ wf,      // bf16 weight fragments
    const float* __restrict__ feat,             // residual input [b][256][HW]
    const float* __restrict__ attn,             // [lvl*2+b][8]
    const float* __restrict__ bns, const float* __restrict__ bnb,
    const float* __restrict__ zpage,            // 256B of zeros (padding source)
    float* __restrict__ out, int lvl, int H, int Wd, int shiftW)
{
    // Double-buffered A tile: [buf(2)][pos(32)][tap(9)][c(32)] bf16 (2x18KB)
    __shared__ __attribute__((aligned(16))) unsigned short lA[2][32 * 9 * 32];
    __shared__ __attribute__((aligned(16))) unsigned short lDummy[1024]; // pad sink
    __shared__ int gPix[320];      // pixel index per (pos,tap), -1 = pad; 288..319 pad

    const int HW    = H * Wd;
    const int b     = blockIdx.y;
    const int Mbase = blockIdx.x * 32;
    const int tid   = threadIdx.x;
    const int lane  = tid & 31;
    const int wv    = tid >> 5;
    const int waveM = wv >> 2;                 // 0..1
    const int waveN = wv & 3;                  // 0..3

    // --- geometry: conv-tap -> input pixel (post-resize), -1 for padding ---
    if (tid < 320) {
        int gp = -1;
        if (tid < 288) {
            int pos = tid / 9, t = tid - pos * 9;
            int p = Mbase + pos;
            int y = p >> shiftW, x = p & (Wd - 1);
            int yy = y + t / 3 - 1, xx = x + (t % 3) - 1;
            if (yy >= 0 && yy < H && xx >= 0 && xx < Wd) gp = yy * Wd + xx;
        }
        gPix[tid] = gp;
    }
    __syncthreads();

    v8f acc[4];
#pragma unroll
    for (int j = 0; j < 4; ++j) acc[j] = (v8f){0.f,0.f,0.f,0.f,0.f,0.f,0.f,0.f};
    v8f cacc[4];
#pragma unroll
    for (int j = 0; j < 4; ++j) cacc[j] = (v8f){0.f,0.f,0.f,0.f,0.f,0.f,0.f,0.f};

    const int mrow = waveM * 16 + (lane & 15);
    const int c8   = ((lane >> 4) & 1) * 8;    // A layout: lanes<16 -> K 0-7/16-23
    const unsigned ldsBase   = (unsigned)(unsigned long long)&lA[0][0];
    const unsigned dummyBase = (unsigned)(unsigned long long)&lDummy[0];
    const unsigned short* resB = res + (size_t)b * HW * ACH;

    // stage tile for flat chunk gc (g = gc>>3, cc = gc&7) into buffer `buf`
    auto stage = [&](int gc, int buf) {
        int g = gc >> 3, cc = gc & 7;
        const unsigned short* chbase = resB + g * 256 + cc * 32;
        unsigned bufbase = ldsBase + (unsigned)(buf * 18432 * 2);
        for (int i = tid; i < 1280; i += 256) {      // exactly 5 issues per thread
            int pair = i >> 2, q = i & 3;
            int pix = gPix[pair];
            const unsigned short* gp = (pix >= 0)
                ? (chbase + (size_t)pix * ACH + q * 8)
                : (const unsigned short*)zpage;
            unsigned ldsoff = (i < 1152) ? (bufbase + (unsigned)(i * 16))
                                         : (dummyBase + (unsigned)((i - 1152) * 16));
            asm volatile("global_load_async_to_lds_b128 %0, %1, off"
                         :: "v"(ldsoff), "v"(gp) : "memory");
        }
    };

    stage(0, 0);                                    // pipeline prologue

    for (int gc = 0; gc < 64; ++gc) {
        int g = gc >> 3, cc = gc & 7, buf = gc & 1;
        if (gc < 63) {
            stage(gc + 1, buf ^ 1);                 // overlap next DMA with compute
            asm volatile("s_wait_asynccnt 5" ::: "memory");   // tile gc landed
        } else {
            asm volatile("s_wait_asynccnt 0" ::: "memory");
        }
        __syncthreads();                            // all waves' tile-gc DMA visible

        // prefetch next weight chunk into L2 (global_prefetch_b8)
        if (gc < 63) {
            int gn = (gc + 1) >> 3, ccn = (gc + 1) & 7;
            size_t blkN = ((((size_t)lvl * 8 + gn) * 9 + 0) * 8 + ccn) * 16 + waveN * 4;
            __builtin_prefetch(wf + blkN * 512 + lane * 16, 0, 1);
        }

        // --- 9 K-steps; one A fragment feeds 4 WMMAs (16M x 64N) ---
        const unsigned short* lbuf = &lA[buf][0];
#pragma unroll
        for (int t = 0; t < 9; ++t) {
            const u16x8* pa = (const u16x8*)&lbuf[mrow * 288 + t * 32 + c8];
            u16x8 lo = pa[0];   // K c8..c8+7
            u16x8 hi = pa[2];   // K c8+16..c8+23
            u16x16 au = __builtin_shufflevector(lo, hi, 0, 1, 2, 3, 4, 5, 6, 7,
                                                8, 9, 10, 11, 12, 13, 14, 15);
            v16bf afr = __builtin_bit_cast(v16bf, au);
#pragma unroll
            for (int j = 0; j < 4; ++j) {
                size_t blk = ((((size_t)lvl * 8 + g) * 9 + t) * 8 + cc) * 16
                             + waveN * 4 + j;
                u16x16 bu = *(const u16x16*)(wf + blk * 512 + lane * 16);
                v16bf bfr = __builtin_bit_cast(v16bf, bu);
                cacc[j] = __builtin_amdgcn_wmma_f32_16x16x32_bf16(
                    false, afr, false, bfr, (short)0, cacc[j], false, false);
            }
        }
        __syncthreads();   // tile-gc consumed: buffer may be refilled at gc+2

        if (cc == 7) {
            // --- per-group epilogue: BN + ReLU + attention weight ---
            float aw = attn[(lvl * 2 + b) * K_CLS + g];
#pragma unroll
            for (int j = 0; j < 4; ++j) {
                int n = waveN * 64 + j * 16 + (lane & 15);
                int o = g * 256 + n;
                float sc = bns[lvl * ACH + o];
                float sh = bnb[lvl * ACH + o];
#pragma unroll
                for (int r = 0; r < 8; ++r) {
                    float v = cacc[j][r] * sc + sh;
                    acc[j][r] += aw * fmaxf(v, 0.f);
                    cacc[j][r] = 0.f;
                }
            }
        }
    }

    // --- final epilogue: residual add + ReLU, store ---
#pragma unroll
    for (int j = 0; j < 4; ++j) {
        int n = waveN * 64 + j * 16 + (lane & 15);
#pragma unroll
        for (int r = 0; r < 8; ++r) {
            int row = Mbase + waveM * 16 + r + ((lane >> 4) << 3);  // M = r + 8*(l/16)
            size_t idx = ((size_t)b * C_INC + n) * HW + row;
            out[idx] = fmaxf(feat[idx] + acc[j][r], 0.f);
        }
    }
}

// ---------------------------------------------------------------------------
extern "C" void kernel_launch(void* const* d_in, const int* in_sizes, int n_in,
                              void* d_out, int out_size, void* d_ws, size_t ws_size,
                              hipStream_t stream) {
    (void)in_sizes; (void)n_in; (void)out_size; (void)ws_size;
    const float* feat[5] = {(const float*)d_in[0], (const float*)d_in[1],
                            (const float*)d_in[2], (const float*)d_in[3],
                            (const float*)d_in[4]};
    const float* attr    = (const float*)d_in[5];
    const float* mlp_f_w = (const float*)d_in[6];
    const float* mlp_f_b = (const float*)d_in[7];
    const float* mlp_a_w = (const float*)d_in[8];
    const float* mlp_a_b = (const float*)d_in[9];
    const float* conv_w  = (const float*)d_in[10];
    const float* bn_g    = (const float*)d_in[11];
    const float* bn_b    = (const float*)d_in[12];
    const float* bn_m    = (const float*)d_in[13];
    const float* bn_v    = (const float*)d_in[14];

    char* ws = (char*)d_ws;
    unsigned short* attrT = (unsigned short*)(ws + WS_ATTRT);
    unsigned short* resP[5] = {
        (unsigned short*)(ws + WS_RES0),
        (unsigned short*)(ws + WS_ATTRT),   // level 1 (64x64) is identity
        (unsigned short*)(ws + WS_RES2),
        (unsigned short*)(ws + WS_RES3),
        (unsigned short*)(ws + WS_RES4)};
    unsigned short* wfrag = (unsigned short*)(ws + WS_WFRAG);
    float* avgattr = (float*)(ws + WS_AVGATTR);
    float* am      = (float*)(ws + WS_AM);
    float* avgx    = (float*)(ws + WS_AVGX);
    float* fx      = (float*)(ws + WS_FX);
    float* attnp   = (float*)(ws + WS_ATTN);
    float* bns     = (float*)(ws + WS_BNS);
    float* bnb     = (float*)(ws + WS_BNB);
    float* zpage   = (float*)(ws + WS_ZERO);
    float* out     = (float*)d_out;

    const int Hs[5] = {128, 64, 32, 16, 8};
    const int shW[5] = {7, 6, 5, 4, 3};

    // precompute / data-movement passes
    k_attr_to_bf16_t<<<65536, 256, 0, stream>>>(attr, attrT);
    for (int lvl = 0; lvl < 5; ++lvl) {
        if (lvl == 1) continue;                       // identity resize
        int H = Hs[lvl], HW = H * H;
        k_resize<<<2 * HW, 256, 0, stream>>>(attrT, resP[lvl], H, H, shW[lvl], HW);
    }
    k_weights_frag<<<92160, 256, 0, stream>>>(conv_w, wfrag);
    k_avg_attr<<<4096, 256, 0, stream>>>(attr, avgattr);
    k_mlp<<<16 * 256, 256, 0, stream>>>(avgattr, mlp_a_w, mlp_a_b, am);
    for (int lvl = 0; lvl < 5; ++lvl) {
        int HW = Hs[lvl] * Hs[lvl];
        k_avg_feat<<<512, 256, 0, stream>>>(feat[lvl], avgx + lvl * 512, HW);
    }
    k_mlp<<<10 * 256, 256, 0, stream>>>(avgx, mlp_f_w, mlp_f_b, fx);
    k_attn<<<10, 256, 0, stream>>>(am, fx, attnp);
    k_bnfold<<<40, 256, 0, stream>>>(bn_g, bn_b, bn_m, bn_v, bns, bnb, zpage);

    // heavy fused conv per level
    size_t outOff = 0;
    for (int lvl = 0; lvl < 5; ++lvl) {
        int H = Hs[lvl], W = Hs[lvl];
        int HW = H * W;
        dim3 grid(HW / 32, 2, 1);
        k_fused_conv<<<grid, 256, 0, stream>>>(resP[lvl], wfrag, feat[lvl], attnp,
                                               bns, bnb, zpage, out + outOff,
                                               lvl, H, W, shW[lvl]);
        outOff += (size_t)2 * C_INC * HW;
    }
}
